// FlashAttentionNaive_17617955848636
// MI455X (gfx1250) — matrix operand order repaired and checked by reading, not compiled, not run on hardware
//
#include <hip/hip_runtime.h>
#include <math.h>

typedef __attribute__((ext_vector_type(16))) _Float16 v16h;
typedef __attribute__((ext_vector_type(8)))  _Float16 v8h;
typedef __attribute__((ext_vector_type(4)))  _Float16 v4h;
typedef __attribute__((ext_vector_type(8)))  float    v8f;
typedef __attribute__((ext_vector_type(4)))  float    v4f;
typedef __attribute__((ext_vector_type(4)))  unsigned int v4u;
typedef __attribute__((ext_vector_type(8)))  int      v8i;
typedef __attribute__((ext_vector_type(4)))  int      v4i;

#define S_LEN 2048
#define DK    128
#define BC    128      // key block
#define QT    128      // query tile per workgroup
#define NW    8        // waves per workgroup (256 threads, wave32)
#define LSTR  136      // padded LDS row stride in f16 elements (272B = 256B data + 16B pad)

union Frag16 { v16h v; v8h h[2]; };

// A-fragment (16x32 f16, M x K) from row-major LDS tile (ISA 7.12.2 layout).
__device__ __forceinline__ v16h ldsA(const _Float16* base, int lane, int kbase) {
    const int m  = lane & 15;
    const int hi = lane >> 4;
    const _Float16* p = base + m * LSTR + kbase + hi * 8;
    Frag16 a;
    a.h[0] = *(const v8h*)(p);        // K = kbase + hi*8 + 0..7
    a.h[1] = *(const v8h*)(p + 16);   // K = kbase + hi*8 + 16..23
    return a.v;
}

// B-fragment (32x16 f16, K x N) where the LDS tile is stored [N][K] row-major.
__device__ __forceinline__ v16h ldsB(const _Float16* base, int lane, int nbase, int kbase) {
    const int n  = nbase + (lane & 15);
    const int hi = lane >> 4;
    const _Float16* p = base + n * LSTR + kbase + hi * 8;
    Frag16 b;
    b.h[0] = *(const v8h*)(p);
    b.h[1] = *(const v8h*)(p + 16);
    return b.v;
}

__device__ __forceinline__ float halfMax(float v) {
    v = fmaxf(v, __shfl_xor(v, 1, 32));
    v = fmaxf(v, __shfl_xor(v, 2, 32));
    v = fmaxf(v, __shfl_xor(v, 4, 32));
    v = fmaxf(v, __shfl_xor(v, 8, 32));
    return v;
}
__device__ __forceinline__ float halfSum(float v) {
    v += __shfl_xor(v, 1, 32);
    v += __shfl_xor(v, 2, 32);
    v += __shfl_xor(v, 4, 32);
    v += __shfl_xor(v, 8, 32);
    return v;
}

// ---------------- Tensor Data Mover (2D tile, f16, with LDS row padding) ----------------
// D# per ISA cdna5 §8.3/§8.4. Pads 4 DWORDs (16B) after every 64 DWORDs (256B) written to
// LDS -> produces the LSTR=136 f16 padded row stride used by the fragment loaders.
__device__ __forceinline__ void make_tdm_2d(unsigned int lds_off, const _Float16* gaddr,
                                            unsigned int tdim0, unsigned int tdim1,
                                            unsigned long long stride0,
                                            unsigned int tile0, unsigned int tile1,
                                            v4u& g0, v8i& g1) {
    const unsigned long long ga = (unsigned long long)gaddr;
    g0.x = 1u;                                          // count=1, user descriptor
    g0.y = lds_off;                                     // LDS byte address
    g0.z = (unsigned int)(ga & 0xFFFFFFFFu);            // global_addr[31:0]
    g0.w = (unsigned int)((ga >> 32) & 0x01FFFFFFu) | (2u << 30);  // addr[56:32] | type=2
    const unsigned int w0 = (1u << 16)   // data_size = 1 -> 2 bytes
                          | (1u << 20)   // pad_enable
                          | (5u << 22)   // pad_interval: 2^(5+1)=64 DWORDs (256B)
                          | (3u << 25);  // pad_amount: 3+1 = 4 DWORDs (16B)
    const unsigned int w1 = (tdim0 & 0xFFFFu) << 16;                       // dim0 lo16
    const unsigned int w2 = (tdim0 >> 16) | ((tdim1 & 0xFFFFu) << 16);     // dim0 hi | dim1 lo
    const unsigned int w3 = (tdim1 >> 16) | (tile0 << 16);                 // dim1 hi | tile_dim0
    const unsigned int w4 = (tile1 & 0xFFFFu);                             // tile_dim1, tile_dim2=0
    const unsigned int w5 = (unsigned int)(stride0 & 0xFFFFFFFFu);         // dim0_stride lo
    const unsigned int w6 = (unsigned int)((stride0 >> 32) & 0xFFFFu);     // dim0_stride hi
    g1 = (v8i){(int)w0, (int)w1, (int)w2, (int)w3, (int)w4, (int)w5, (int)w6, 0};
}

__device__ __forceinline__ void tdm_load(v4u g0, v8i g1) {
    v4i gz = {0, 0, 0, 0};
#if __has_include(<hip/amd_detail/amd_gfx1250_TDM.h>)
    v8i gz8 = {0, 0, 0, 0, 0, 0, 0, 0};
    __builtin_amdgcn_tensor_load_to_lds(g0, g1, gz, gz, gz8, 0);   // clang-23 6-arg form
#else
    __builtin_amdgcn_tensor_load_to_lds(g0, g1, gz, gz, 0);        // ROCm 7.2 5-arg form
#endif
}

// ---------------- pre-pass: K -> f16 row-major, V -> f16 transposed [d][s] ----------------
__global__ __launch_bounds__(256)
void cvt_kv_kernel(const float* __restrict__ K, const float* __restrict__ V,
                   _Float16* __restrict__ Kh, _Float16* __restrict__ Vt) {
    const int bh = blockIdx.y;
    const size_t base = (size_t)bh * S_LEN * DK;
    const int e4 = blockIdx.x * 256 + threadIdx.x;     // one float4 per thread
    const int s  = e4 / (DK / 4);
    const int d4 = (e4 % (DK / 4)) * 4;

    v4f k4 = *(const v4f*)(K + base + (size_t)s * DK + d4);
    v4h kh; kh.x = (_Float16)k4.x; kh.y = (_Float16)k4.y;
            kh.z = (_Float16)k4.z; kh.w = (_Float16)k4.w;
    *(v4h*)(Kh + base + (size_t)s * DK + d4) = kh;

    v4f vv = *(const v4f*)(V + base + (size_t)s * DK + d4);
    _Float16* vt = Vt + (size_t)bh * DK * S_LEN;
    vt[(size_t)(d4 + 0) * S_LEN + s] = (_Float16)vv.x;
    vt[(size_t)(d4 + 1) * S_LEN + s] = (_Float16)vv.y;
    vt[(size_t)(d4 + 2) * S_LEN + s] = (_Float16)vv.z;
    vt[(size_t)(d4 + 3) * S_LEN + s] = (_Float16)vv.w;
}

// ---------------- main flash-attention kernel (TDM double-buffered) ----------------
__global__ __launch_bounds__(256, 1)
void fa_fwd_wmma(const float* __restrict__ Qg, const _Float16* __restrict__ Kh,
                 const _Float16* __restrict__ Vth, float* __restrict__ Og) {
    extern __shared__ _Float16 smem[];
    _Float16* Qs  = smem;                   // QT x LSTR (scaled q, f16)
    _Float16* KsB = Qs + QT * LSTR;         // 2 x BC x LSTR ([key][d]) -- TDM dst
    _Float16* VtB = KsB + 2 * BC * LSTR;    // 2 x DK x LSTR ([d][key]) -- TDM dst
    _Float16* Ps  = VtB + 2 * DK * LSTR;    // NW*16 x LSTR (per-wave P scratch)

    const int tid  = threadIdx.x;
    const int lane = tid & 31;
    const int wave = tid >> 5;
    const int hi   = lane >> 4;
    const int l15  = lane & 15;

    const int qt = blockIdx.x;
    const int bh = blockIdx.y;
    const int q0 = qt * QT;

    const float scale = 0.08838834764831845f;   // 1/sqrt(128)

    // stage Q tile (pre-scaled) into LDS as f16
    const float* Qp = Qg + ((size_t)bh * S_LEN + q0) * DK;
    for (int idx = tid; idx < QT * DK / 4; idx += 256) {
        const int r  = idx / (DK / 4);
        const int c4 = (idx % (DK / 4)) * 4;
        v4f q4 = *(const v4f*)(Qp + r * DK + c4);
        v4h qh; qh.x = (_Float16)(q4.x * scale); qh.y = (_Float16)(q4.y * scale);
                qh.z = (_Float16)(q4.z * scale); qh.w = (_Float16)(q4.w * scale);
        *(v4h*)(Qs + r * LSTR + c4) = qh;
    }

    const _Float16* Khead = Kh  + (size_t)bh * S_LEN * DK;
    const _Float16* Vhead = Vth + (size_t)bh * DK * S_LEN;
    unsigned int ks_off[2], vt_off[2];
    ks_off[0] = (unsigned int)(unsigned long long)(void*)(KsB);
    ks_off[1] = (unsigned int)(unsigned long long)(void*)(KsB + BC * LSTR);
    vt_off[0] = (unsigned int)(unsigned long long)(void*)(VtB);
    vt_off[1] = (unsigned int)(unsigned long long)(void*)(VtB + DK * LSTR);

    v8f   Oacc[8];
    float mrow[8], lrow[8];
#pragma unroll
    for (int t = 0; t < 8; ++t)
#pragma unroll
        for (int r = 0; r < 8; ++r) Oacc[t][r] = 0.0f;
#pragma unroll
    for (int r = 0; r < 8; ++r) { mrow[r] = -INFINITY; lrow[r] = 0.0f; }

    const int jmax = qt;   // blocks j > qt are exact no-ops under the causal recurrence

    // prologue: issue DMA for block 0 into buffer 0
    if (wave == 0) {
        v4u g0; v8i g1;
        make_tdm_2d(ks_off[0], Khead, DK, BC, DK, DK, BC, g0, g1);
        tdm_load(g0, g1);
        make_tdm_2d(vt_off[0], Vhead, S_LEN, DK, S_LEN, BC, DK, g0, g1);
        tdm_load(g0, g1);
    }

    for (int j = 0; j <= jmax; ++j) {
        const int buf = j & 1;
        if (wave == 0) __builtin_amdgcn_s_wait_tensorcnt(0);   // buffer `buf` arrived
        __syncthreads();   // all waves: data ready AND done reading buffer buf^1

        // pipeline: kick off DMA for block j+1 into the other buffer
        if (wave == 0 && j < jmax) {
            v4u g0; v8i g1;
            make_tdm_2d(ks_off[buf ^ 1], Khead + (size_t)(j + 1) * BC * DK,
                        DK, BC, DK, DK, BC, g0, g1);
            tdm_load(g0, g1);
            make_tdm_2d(vt_off[buf ^ 1], Vhead + (size_t)(j + 1) * BC,
                        S_LEN, DK, S_LEN, BC, DK, g0, g1);
            tdm_load(g0, g1);
        }

        const _Float16* Ks  = KsB + buf * (BC * LSTR);
        const _Float16* Vts = VtB + buf * (DK * LSTR);

        // ---- S = (scaled Q rows) x K^T : 8 N-tiles x 4 K-steps ----
        const _Float16* Qw = Qs + (wave * 16) * LSTR;
        v16h aQ[4];
#pragma unroll
        for (int kk = 0; kk < 4; ++kk) aQ[kk] = ldsA(Qw, lane, kk * 32);

        v8f Sacc[8];
#pragma unroll
        for (int t = 0; t < 8; ++t) {
#pragma unroll
            for (int r = 0; r < 8; ++r) Sacc[t][r] = 0.0f;
#pragma unroll
            for (int kk = 0; kk < 4; ++kk) {
                v16h b = ldsB(Ks, lane, t * 16, kk * 32);
                Sacc[t] = __builtin_amdgcn_wmma_f32_16x16x32_f16(
                    false, aQ[kk], false, b, (short)0, Sacc[t], false, false);
            }
        }

        // ---- causal mask (diagonal block only) ----
        if (j == qt) {
            const int rbase = wave * 16 + hi * 8;
#pragma unroll
            for (int t = 0; t < 8; ++t) {
                const int n = t * 16 + l15;
#pragma unroll
                for (int r = 0; r < 8; ++r)
                    if (n > rbase + r) Sacc[t][r] = -INFINITY;
            }
        }

        // ---- online softmax (faithful to reference recurrence) ----
        float fold[8], invl[8], eijr[8];
#pragma unroll
        for (int r = 0; r < 8; ++r) {
            float mv = Sacc[0][r];
#pragma unroll
            for (int t = 1; t < 8; ++t) mv = fmaxf(mv, Sacc[t][r]);
            const float mij  = halfMax(mv);
            const float mnew = fmaxf(mrow[r], mij);
            float psum = 0.0f;
#pragma unroll
            for (int t = 0; t < 8; ++t) {
                const float p = __expf(Sacc[t][r] - mnew);
                Sacc[t][r] = p;
                psum += p;
            }
            psum = halfSum(psum);
            const float ei   = __expf(mrow[r] - mnew);
            const float eij  = __expf(mij   - mnew);
            const float lnew = ei * lrow[r] + psum;
            fold[r] = ei * lrow[r];
            eijr[r] = eij;
            invl[r] = __builtin_amdgcn_rcpf(lnew);   // fast v_rcp_f32, no IEEE div chain
            mrow[r] = mnew;
            lrow[r] = lnew;
        }

        // ---- spill (e_ij * P) to per-wave LDS scratch as f16 ----
        _Float16* Pw = Ps + wave * 16 * LSTR;
#pragma unroll
        for (int t = 0; t < 8; ++t) {
            const int n = t * 16 + l15;
#pragma unroll
            for (int r = 0; r < 8; ++r)
                Pw[(hi * 8 + r) * LSTR + n] = (_Float16)(Sacc[t][r] * eijr[r]);
        }

        // ---- O = (e_i*l*O + (e_ij*P) x V) / l_new ----
        v16h aP[4];
#pragma unroll
        for (int kk = 0; kk < 4; ++kk) aP[kk] = ldsA(Pw, lane, kk * 32);
#pragma unroll
        for (int t = 0; t < 8; ++t) {
            v8f pv;
#pragma unroll
            for (int r = 0; r < 8; ++r) pv[r] = 0.0f;
#pragma unroll
            for (int kk = 0; kk < 4; ++kk) {
                v16h b = ldsB(Vts, lane, t * 16, kk * 32);
                pv = __builtin_amdgcn_wmma_f32_16x16x32_f16(
                    false, aP[kk], false, b, (short)0, pv, false, false);
            }
#pragma unroll
            for (int r = 0; r < 8; ++r)
                Oacc[t][r] = (fold[r] * Oacc[t][r] + pv[r]) * invl[r];
        }
    }

    // ---- write O (fp32) ----
    float* Op = Og + ((size_t)bh * S_LEN + q0 + wave * 16) * DK;
#pragma unroll
    for (int t = 0; t < 8; ++t) {
        const int n = t * 16 + l15;
#pragma unroll
        for (int r = 0; r < 8; ++r)
            Op[(hi * 8 + r) * DK + n] = Oacc[t][r];
    }
}

extern "C" void kernel_launch(void* const* d_in, const int* in_sizes, int n_in,
                              void* d_out, int out_size, void* d_ws, size_t ws_size,
                              hipStream_t stream) {
    const float* Q = (const float*)d_in[0];
    const float* K = (const float*)d_in[1];
    const float* V = (const float*)d_in[2];
    float* O = (float*)d_out;

    // f16 workspace: K (row-major) + V (transposed), 16 MB each
    _Float16* Kh  = (_Float16*)d_ws;
    _Float16* Vth = Kh + (size_t)2 * 16 * S_LEN * DK;

    dim3 cgrid(S_LEN * DK / 4 / 256, 2 * 16);
    cvt_kv_kernel<<<cgrid, 256, 0, stream>>>(K, V, Kh, Vth);

    // LDS: Q + 2x(K,Vt) double buffer + P scratch = 204 KB (< 320 KB per WGP)
    const size_t shmem = (size_t)(QT + 2 * BC + 2 * DK + NW * 16) * LSTR * sizeof(_Float16);
    hipFuncSetAttribute(reinterpret_cast<const void*>(fa_fwd_wmma),
                        hipFuncAttributeMaxDynamicSharedMemorySize, (int)shmem);

    dim3 grid(S_LEN / QT, 2 * 16);
    fa_fwd_wmma<<<grid, 256, shmem, stream>>>(Q, Kh, Vth, O);
}
